// LlamaAttention_75977971466862
// MI455X (gfx1250) — matrix operand order repaired and checked by reading, not compiled, not run on hardware
//
#include <hip/hip_runtime.h>
#include <cstdint>
#include <cstddef>

// ---------------------------------------------------------------------------
// LlamaAttention forward for MI455X (gfx1250, wave32, WMMA).
// Pipeline: fp32->bf16 converts -> QKV GEMMs (fused RoPE+RMSNorm epilogue)
//           -> flash attention (online softmax, bf16 WMMA) -> out-proj GEMM.
// Data movement: GLOBAL_LOAD_ASYNC_TO_LDS_B128 double-buffering (ASYNCcnt),
//                global_prefetch warming L2 two chunks ahead.
// ---------------------------------------------------------------------------

#define S_   2048
#define D_   2048
#define NH_  32
#define NKV_ 8
#define HD_  64
#define NREP_ 4
#define EPS_ 1e-5f
#define INV_SCALE_ 0.125f   // 1/sqrt(64)

typedef __bf16 bf16;
typedef bf16  v16bf __attribute__((ext_vector_type(16)));
typedef bf16  v8bf  __attribute__((ext_vector_type(8)));
typedef float v8f   __attribute__((ext_vector_type(8)));
typedef int   v4i_  __attribute__((vector_size(4 * sizeof(int))));

static __device__ __forceinline__ bf16 f2bf(float x) { return (bf16)x; }
static __device__ __forceinline__ unsigned short f2bf_bits(float x) {
  union { bf16 h; unsigned short s; } c; c.h = (bf16)x; return c.s;
}

static __device__ __forceinline__ v8f wmma_bf16(v16bf a, v16bf b, v8f c) {
  // (neg_a, A, neg_b, B, c_mod, C, reuse_a, reuse_b)
  return __builtin_amdgcn_wmma_f32_16x16x32_bf16(false, a, false, b,
                                                 (short)0, c, false, false);
}

// Build a 16-element bf16 fragment from two 16B-aligned LDS runs.
static __device__ __forceinline__ v16bf ldfrag(const bf16* p0, const bf16* p1) {
  v8bf lo = *(const v8bf*)p0;
  v8bf hi = *(const v8bf*)p1;
  return __builtin_shufflevector(lo, hi, 0,1,2,3,4,5,6,7,8,9,10,11,12,13,14,15);
}

// ---------------------------------------------------------------------------
// Async global->LDS copy (16B per lane). Falls back to a synchronous copy if
// the toolchain lacks the gfx1250 async builtin.
// Builtin signature (per clang diagnostic): (global int4*, local int4*, i32, i32)
// ---------------------------------------------------------------------------
static __device__ __forceinline__ void cp_async16(const bf16* g, bf16* l) {
#if __has_builtin(__builtin_amdgcn_global_load_async_to_lds_b128)
  __builtin_amdgcn_global_load_async_to_lds_b128(
      (__attribute__((address_space(1))) v4i_*)(g),
      (__attribute__((address_space(3))) v4i_*)(l), 0, 0);
#else
  *(uint4*)l = *(const uint4*)g;
#endif
}
static __device__ __forceinline__ void cp_async_wait() {
#if __has_builtin(__builtin_amdgcn_global_load_async_to_lds_b128)
#if __has_builtin(__builtin_amdgcn_s_wait_asynccnt)
  __builtin_amdgcn_s_wait_asynccnt(0);
#else
  asm volatile("s_wait_asynccnt 0" ::: "memory");
#endif
#endif
}

// ---------------------------------------------------------------------------
// fp32 -> bf16 bulk convert (vectorized 4-wide)
// ---------------------------------------------------------------------------
__global__ __launch_bounds__(256)
void cvt_bf16_kernel(const float4* __restrict__ src, ushort4* __restrict__ dst,
                     int n4) {
  int i = blockIdx.x * blockDim.x + threadIdx.x;
  if (i < n4) {
    float4 v = src[i];
    ushort4 o;
    o.x = f2bf_bits(v.x);
    o.y = f2bf_bits(v.y);
    o.z = f2bf_bits(v.z);
    o.w = f2bf_bits(v.w);
    dst[i] = o;
  }
}

// ---------------------------------------------------------------------------
// Tiled GEMM: C[m0:m0+128, n0:n0+64] = A(bf16, SxD) @ W(bf16, NxD)^T
// MODE 0: Q proj  -> RoPE+RMSNorm -> bf16 q[s][h*64+d]
// MODE 1: K proj  -> RoPE+RMSNorm -> bf16 k[g][s][d] + fp32 kh_t[h][d][s] (x4)
// MODE 2: V proj  ->              -> bf16 v[g][s][d] + fp32 vh[h][s][d]   (x4)
// MODE 3: out-proj ->             -> fp32 y[s][n]
// 256 threads = 8 waves; wave (wm,wn) owns a 32x32 subtile (2x2 WMMA frags).
// K-chunks of 64 double-buffered in LDS via async DMA.
// ---------------------------------------------------------------------------
template<int MODE>
__global__ __launch_bounds__(256)
void gemm_kernel(const bf16* __restrict__ A, const bf16* __restrict__ W,
                 const float* __restrict__ fcos, const float* __restrict__ fsin,
                 const float* __restrict__ normw,
                 bf16* __restrict__ out_bf, float* __restrict__ out_f32) {
  __shared__ __align__(16) bf16 As[2][128][72];
  __shared__ __align__(16) bf16 Bs[2][64][72];

  const int tid = threadIdx.x;
  const int m0  = blockIdx.x * 128;
  const int nt  = blockIdx.y;        // head index (or 64-col tile for MODE 3)
  const int n0  = nt * 64;

  const int w  = tid >> 5;
  const int ln = tid & 31;
  const int lh = ln >> 4;            // half-wave
  const int ll = ln & 15;
  const int wm = w & 3;              // 4 waves along M
  const int wn = w >> 2;             // 2 waves along N

  v8f c[2][2] = {};

  auto issue_chunk = [&](int kc, int buf) {
#pragma unroll
    for (int i = 0; i < 4; ++i) {
      int seg = tid + i * 256;
      int row = seg >> 3, ks = (seg & 7) * 8;
      cp_async16(A + (size_t)(m0 + row) * D_ + kc + ks, &As[buf][row][ks]);
    }
#pragma unroll
    for (int i = 0; i < 2; ++i) {
      int seg = tid + i * 256;
      int row = seg >> 3, ks = (seg & 7) * 8;
      cp_async16(W + (size_t)(n0 + row) * D_ + kc + ks, &Bs[buf][row][ks]);
    }
  };

  issue_chunk(0, 0);
  cp_async_wait();
  __syncthreads();

  const int chunks = D_ / 64;        // 32
  for (int cidx = 0; cidx < chunks; ++cidx) {
    const int buf = cidx & 1;
    if (cidx + 1 < chunks) issue_chunk((cidx + 1) * 64, buf ^ 1);
    if (cidx + 2 < chunks) {         // warm L2 two chunks ahead
      const int kc2 = (cidx + 2) * 64;
      __builtin_prefetch(A + (size_t)(m0 + (tid >> 1)) * D_ + kc2 + (tid & 1) * 32, 0, 1);
      __builtin_prefetch(W + (size_t)(n0 + ((tid >> 2) & 63)) * D_ + kc2 + (tid & 3) * 16, 0, 1);
    }

    // ---- compute: 2 K-steps x (2x2) WMMA ----
#pragma unroll
    for (int kstep = 0; kstep < 2; ++kstep) {
      const int ka = kstep * 32 + lh * 8;   // A frag: K {base..+7, base+16..+23}
      const int kb = kstep * 32 + lh * 8;   // B frag: K {base..+7, base+16..+23}
      v16bf a0 = ldfrag(&As[buf][wm * 32 +      ll][ka], &As[buf][wm * 32 +      ll][ka + 16]);
      v16bf a1 = ldfrag(&As[buf][wm * 32 + 16 + ll][ka], &As[buf][wm * 32 + 16 + ll][ka + 16]);
      v16bf b0 = ldfrag(&Bs[buf][wn * 32 +      ll][kb], &Bs[buf][wn * 32 +      ll][kb + 16]);
      v16bf b1 = ldfrag(&Bs[buf][wn * 32 + 16 + ll][kb], &Bs[buf][wn * 32 + 16 + ll][kb + 16]);
      c[0][0] = wmma_bf16(a0, b0, c[0][0]);
      c[0][1] = wmma_bf16(a0, b1, c[0][1]);
      c[1][0] = wmma_bf16(a1, b0, c[1][0]);
      c[1][1] = wmma_bf16(a1, b1, c[1][1]);
    }

    cp_async_wait();                 // next chunk landed in LDS
    __syncthreads();
  }

  // ---- epilogue ----
  if constexpr (MODE == 3) {
#pragma unroll
    for (int sm = 0; sm < 2; ++sm)
#pragma unroll
      for (int sn = 0; sn < 2; ++sn)
#pragma unroll
        for (int r = 0; r < 8; ++r) {
          int row = wm * 32 + sm * 16 + r + 8 * lh;
          int col = wn * 32 + sn * 16 + ll;
          out_f32[(size_t)(m0 + row) * D_ + n0 + col] = c[sm][sn][r];
        }
  } else {
    __shared__ __align__(16) float Cs[128][66];
#pragma unroll
    for (int sm = 0; sm < 2; ++sm)
#pragma unroll
      for (int sn = 0; sn < 2; ++sn)
#pragma unroll
        for (int r = 0; r < 8; ++r) {
          int row = wm * 32 + sm * 16 + r + 8 * lh;
          int col = wn * 32 + sn * 16 + ll;
          Cs[row][col] = c[sm][sn][r];
        }
    __syncthreads();

    if (tid < 128) {
      const int s = m0 + tid;
      float x[64];
#pragma unroll
      for (int d = 0; d < 64; ++d) x[d] = Cs[tid][d];

      if constexpr (MODE <= 1) {
        // RoPE (interleaved pairs), then RMSNorm over the 64-dim head vector
#pragma unroll
        for (int j = 0; j < 32; ++j) {
          float cc = fcos[(size_t)s * 32 + j];
          float ss = fsin[(size_t)s * 32 + j];
          float xr = x[2 * j], xi = x[2 * j + 1];
          x[2 * j]     = xr * cc - xi * ss;
          x[2 * j + 1] = xr * ss + xi * cc;
        }
        float acc = 0.0f;
#pragma unroll
        for (int d = 0; d < 64; ++d) acc += x[d] * x[d];
        float inv = rsqrtf(acc * (1.0f / 64.0f) + EPS_);
#pragma unroll
        for (int d = 0; d < 64; ++d) x[d] = x[d] * inv * normw[d];
      }

      if constexpr (MODE == 0) {               // Q: bf16 [s][h*64+d]
#pragma unroll
        for (int d = 0; d < 64; ++d)
          out_bf[(size_t)s * D_ + nt * 64 + d] = f2bf(x[d]);
      } else if constexpr (MODE == 1) {        // K
#pragma unroll
        for (int d = 0; d < 64; ++d)
          out_bf[((size_t)nt * S_ + s) * HD_ + d] = f2bf(x[d]);
#pragma unroll
        for (int rep = 0; rep < NREP_; ++rep) {
          int h = nt * NREP_ + rep;            // kh_t[h][d][s] (fp32)
#pragma unroll
          for (int d = 0; d < 64; ++d)
            out_f32[((size_t)(h * HD_ + d)) * S_ + s] = x[d];
        }
      } else {                                 // V
#pragma unroll
        for (int d = 0; d < 64; ++d)
          out_bf[((size_t)nt * S_ + s) * HD_ + d] = f2bf(x[d]);
#pragma unroll
        for (int rep = 0; rep < NREP_; ++rep) {
          int h = nt * NREP_ + rep;            // vh[h][s][d] (fp32)
#pragma unroll
          for (int d = 0; d < 64; ++d)
            out_f32[((size_t)h * S_ + s) * HD_ + d] = x[d];
        }
      }
    }
  }
}

// ---------------------------------------------------------------------------
// Flash attention: one (head, 64-row q block) per workgroup, 4 waves.
// Wave w owns q rows [q0 + 16w, +16). Online softmax over 64-key blocks.
// ---------------------------------------------------------------------------
__global__ __launch_bounds__(128)
void attn_kernel(const bf16* __restrict__ Q, const bf16* __restrict__ K,
                 const bf16* __restrict__ V, bf16* __restrict__ Y) {
  __shared__ __align__(16) bf16 Qs[64][72];
  __shared__ __align__(16) bf16 Ks[64][72];       // [key][d]
  __shared__ __align__(16) bf16 Vt[64][72];       // [d][key]
  __shared__ __align__(16) bf16 Ps[4][16][72];    // per-wave P relayout

  const int tid = threadIdx.x;
  const int q0  = blockIdx.x * 64;
  const int h   = blockIdx.y;
  const int g   = h / NREP_;
  const int w   = tid >> 5, ln = tid & 31, lh = ln >> 4, ll = ln & 15;

  // Q tile 64x64 (async DMA; consumed after first barrier pair)
#pragma unroll
  for (int i = 0; i < 4; ++i) {
    int seg = tid + i * 128;
    int row = seg >> 3, ds = (seg & 7) * 8;
    cp_async16(Q + (size_t)(q0 + row) * D_ + h * HD_ + ds, &Qs[row][ds]);
  }

  v8f   o[4] = {};
  float mrow[8], lrow[8];
#pragma unroll
  for (int r = 0; r < 8; ++r) { mrow[r] = -1e30f; lrow[r] = 0.0f; }

  const int nblk = blockIdx.x + 1;   // causal: keys up to q0+63
  for (int kb = 0; kb < nblk; ++kb) {
    __syncthreads();                 // safe to overwrite Ks/Vt
#pragma unroll
    for (int i = 0; i < 4; ++i) {
      int seg = tid + i * 128;
      int key = seg >> 3, ds = (seg & 7) * 8;
      cp_async16(K + ((size_t)g * S_ + kb * 64 + key) * HD_ + ds, &Ks[key][ds]);
      uint4 vv = *(const uint4*)(V + ((size_t)g * S_ + kb * 64 + key) * HD_ + ds);
      union { uint4 u; v8bf b; } e; e.u = vv;
#pragma unroll
      for (int j = 0; j < 8; ++j) Vt[ds + j][key] = e.b[j];   // transpose V
    }
    cp_async_wait();
    __syncthreads();

    // S = Q @ K^T : 16x64 per wave, 8 WMMA
    v8f sf[4] = {};
#pragma unroll
    for (int kstep = 0; kstep < 2; ++kstep) {
      const int ka = kstep * 32 + lh * 8;
      const int kd = kstep * 32 + lh * 8;
      v16bf a = ldfrag(&Qs[w * 16 + ll][ka], &Qs[w * 16 + ll][ka + 16]);
#pragma unroll
      for (int nf = 0; nf < 4; ++nf) {
        v16bf b = ldfrag(&Ks[nf * 16 + ll][kd], &Ks[nf * 16 + ll][kd + 16]);
        sf[nf] = wmma_bf16(a, b, sf[nf]);
      }
    }

    // scale + causal mask
#pragma unroll
    for (int nf = 0; nf < 4; ++nf)
#pragma unroll
      for (int r = 0; r < 8; ++r) {
        int row = q0 + w * 16 + r + 8 * lh;
        int col = kb * 64 + nf * 16 + ll;
        float sv = sf[nf][r] * INV_SCALE_;
        sf[nf][r] = (col > row) ? -1e30f : sv;
      }

    // online softmax update (row reductions across 16 lanes of the half-wave)
    float corr[8];
#pragma unroll
    for (int r = 0; r < 8; ++r) {
      float mv = mrow[r];
#pragma unroll
      for (int nf = 0; nf < 4; ++nf) mv = fmaxf(mv, sf[nf][r]);
#pragma unroll
      for (int off = 8; off; off >>= 1) mv = fmaxf(mv, __shfl_xor(mv, off, 16));
      corr[r] = __expf(mrow[r] - mv);
      mrow[r] = mv;
      float sum = 0.0f;
#pragma unroll
      for (int nf = 0; nf < 4; ++nf) {
        float p = __expf(sf[nf][r] - mv);
        sf[nf][r] = p;
        sum += p;
      }
#pragma unroll
      for (int off = 8; off; off >>= 1) sum += __shfl_xor(sum, off, 16);
      lrow[r] = lrow[r] * corr[r] + sum;
    }
#pragma unroll
    for (int nf = 0; nf < 4; ++nf)
#pragma unroll
      for (int r = 0; r < 8; ++r) o[nf][r] *= corr[r];

    // relayout P (C layout -> A layout) through per-wave LDS
#pragma unroll
    for (int nf = 0; nf < 4; ++nf)
#pragma unroll
      for (int r = 0; r < 8; ++r)
        Ps[w][r + 8 * lh][nf * 16 + ll] = f2bf(sf[nf][r]);
    __syncthreads();

    // O += P @ V : 8 WMMA
#pragma unroll
    for (int kstep = 0; kstep < 2; ++kstep) {
      const int ka = kstep * 32 + lh * 8;
      const int kd = kstep * 32 + lh * 8;
      v16bf a = ldfrag(&Ps[w][ll][ka], &Ps[w][ll][ka + 16]);
#pragma unroll
      for (int nf = 0; nf < 4; ++nf) {
        v16bf b = ldfrag(&Vt[nf * 16 + ll][kd], &Vt[nf * 16 + ll][kd + 16]);
        o[nf] = wmma_bf16(a, b, o[nf]);
      }
    }
  }

  // normalize and emit bf16 y_heads[s][h*64+d]
#pragma unroll
  for (int nf = 0; nf < 4; ++nf)
#pragma unroll
    for (int r = 0; r < 8; ++r) {
      int row = q0 + w * 16 + r + 8 * lh;
      int col = h * HD_ + nf * 16 + ll;
      Y[(size_t)row * D_ + col] = f2bf(o[nf][r] * (1.0f / lrow[r]));
    }
}

// ---------------------------------------------------------------------------
extern "C" void kernel_launch(void* const* d_in, const int* in_sizes, int n_in,
                              void* d_out, int out_size, void* d_ws, size_t ws_size,
                              hipStream_t stream) {
  const float* hs   = (const float*)d_in[0];
  const float* fcos = (const float*)d_in[1];
  const float* fsin = (const float*)d_in[2];
  // d_in[3]: atten_mask (unused; causal mask computed analytically)
  const float* wq   = (const float*)d_in[4];
  const float* wk   = (const float*)d_in[5];
  const float* wv   = (const float*)d_in[6];
  const float* wo   = (const float*)d_in[7];
  const float* qn   = (const float*)d_in[8];
  const float* kn   = (const float*)d_in[9];

  float* out    = (float*)d_out;
  float* y_out  = out;                              // [s][d]
  float* kh_out = out + (size_t)S_ * D_;            // [h][hd][s]
  float* vh_out = out + 2 * (size_t)S_ * D_;        // [h][s][hd]

  char* ws = (char*)d_ws;
  bf16* hs_b = (bf16*)ws;  ws += (size_t)S_ * D_ * 2;
  bf16* wq_b = (bf16*)ws;  ws += (size_t)D_ * D_ * 2;
  bf16* wk_b = (bf16*)ws;  ws += (size_t)NKV_ * HD_ * D_ * 2;
  bf16* wv_b = (bf16*)ws;  ws += (size_t)NKV_ * HD_ * D_ * 2;
  bf16* wo_b = (bf16*)ws;  ws += (size_t)D_ * D_ * 2;
  bf16* q_b  = (bf16*)ws;  ws += (size_t)S_ * D_ * 2;
  bf16* k_b  = (bf16*)ws;  ws += (size_t)NKV_ * S_ * HD_ * 2;
  bf16* v_b  = (bf16*)ws;  ws += (size_t)NKV_ * S_ * HD_ * 2;
  bf16* y_b  = (bf16*)ws;  ws += (size_t)S_ * D_ * 2;

  auto cvt = [&](const float* src, bf16* dst, size_t n) {
    int n4 = (int)(n / 4);
    cvt_bf16_kernel<<<(n4 + 255) / 256, 256, 0, stream>>>(
        (const float4*)src, (ushort4*)dst, n4);
  };
  cvt(hs, hs_b, (size_t)S_ * D_);
  cvt(wq, wq_b, (size_t)D_ * D_);
  cvt(wk, wk_b, (size_t)NKV_ * HD_ * D_);
  cvt(wv, wv_b, (size_t)NKV_ * HD_ * D_);
  cvt(wo, wo_b, (size_t)D_ * D_);

  gemm_kernel<0><<<dim3(S_ / 128, NH_),  256, 0, stream>>>(hs_b, wq_b, fcos, fsin, qn, q_b, nullptr);
  gemm_kernel<1><<<dim3(S_ / 128, NKV_), 256, 0, stream>>>(hs_b, wk_b, fcos, fsin, kn, k_b, kh_out);
  gemm_kernel<2><<<dim3(S_ / 128, NKV_), 256, 0, stream>>>(hs_b, wv_b, nullptr, nullptr, nullptr, v_b, vh_out);

  attn_kernel<<<dim3(S_ / 64, NH_), 128, 0, stream>>>(q_b, k_b, v_b, y_b);

  gemm_kernel<3><<<dim3(S_ / 128, D_ / 64), 256, 0, stream>>>(y_b, wo_b, nullptr, nullptr, nullptr, nullptr, y_out);
}
